// AttentionReweightingFusion_60524679135569
// MI455X (gfx1250) — matrix operand order repaired and checked by reading, not compiled, not run on hardware
//
#include <hip/hip_runtime.h>
#include <hip/hip_bf16.h>

typedef __attribute__((ext_vector_type(16))) __bf16 v16bf;
typedef __attribute__((ext_vector_type(8)))  __bf16 v8bf;
typedef __attribute__((ext_vector_type(8)))  float  v8f;

#define HD 512

static __device__ __forceinline__ float geluf(float x) {
    return 0.5f * x * (1.0f + erff(x * 0.70710678118654752f));
}
static __device__ __forceinline__ float sigm(float x) {
    return 1.0f / (1.0f + __expf(-x));
}
// round-to-nearest-even float -> bf16 bits
static __device__ __forceinline__ unsigned short f2bf(float f) {
    unsigned int u = __float_as_uint(f);
    u += 0x7FFFu + ((u >> 16) & 1u);
    return (unsigned short)(u >> 16);
}

// ---------------------------------------------------------------------------
// Repack a KxN f32 weight matrix into WMMA B-fragment-friendly bf16 layout:
// frag (kt, nt): 32 lanes x 16 contiguous bf16. lane = half*16 + n holds
// elements e=0..15  <->  W[kt*32 + half*16 + e][nt*16 + n].
// ---------------------------------------------------------------------------
__global__ void pack_weight(const float* __restrict__ W, unsigned short* __restrict__ P,
                            int K, int N) {
    int idx = blockIdx.x * blockDim.x + threadIdx.x;
    int total = (K / 32) * (N / 16) * 32;
    if (idx >= total) return;
    int lane = idx & 31;
    int t    = idx >> 5;
    int nt   = t % (N / 16);
    int kt   = t / (N / 16);
    int half = lane >> 4;
    int n    = lane & 15;
    int kbase = kt * 32 + half * 16;
    int col   = nt * 16 + n;
#pragma unroll
    for (int e = 0; e < 16; ++e)
        P[(size_t)idx * 16 + e] = f2bf(W[(size_t)(kbase + e) * N + col]);
}

// ---------------------------------------------------------------------------
// Tiny per-row MLPs: quality-attention (11->64->32->1, gelu,gelu,sigmoid) and
// modality-importance (4->32->2, gelu, softmax). Emits wImg = img_w*q_att,
// wText = text_w*q_att.
// ---------------------------------------------------------------------------
__global__ void rowprep(const float* __restrict__ quality,
                        const float* __restrict__ qa_w1, const float* __restrict__ qa_b1,
                        const float* __restrict__ qa_w2, const float* __restrict__ qa_b2,
                        const float* __restrict__ qa_w3, const float* __restrict__ qa_b3,
                        const float* __restrict__ mi_w1, const float* __restrict__ mi_b1,
                        const float* __restrict__ mi_w2, const float* __restrict__ mi_b2,
                        float* __restrict__ wImg, float* __restrict__ wText, int B) {
    int r = blockIdx.x * blockDim.x + threadIdx.x;
    if (r >= B) return;
    float q[11];
#pragma unroll
    for (int j = 0; j < 11; ++j) q[j] = quality[(size_t)r * 11 + j];

    float h1[64];
#pragma unroll
    for (int o = 0; o < 64; ++o) {
        float s = qa_b1[o];
#pragma unroll
        for (int j = 0; j < 11; ++j) s += q[j] * qa_w1[j * 64 + o];
        h1[o] = geluf(s);
    }
    float h2[32];
#pragma unroll
    for (int o = 0; o < 32; ++o) {
        float s = qa_b2[o];
#pragma unroll
        for (int j = 0; j < 64; ++j) s += h1[j] * qa_w2[j * 32 + o];
        h2[o] = geluf(s);
    }
    float s3 = qa_b3[0];
#pragma unroll
    for (int j = 0; j < 32; ++j) s3 += h2[j] * qa_w3[j];
    float qatt = sigm(s3);

    float ia[4] = {q[6], q[7], q[8], q[9]};
    float m1[32];
#pragma unroll
    for (int o = 0; o < 32; ++o) {
        float s = mi_b1[o];
#pragma unroll
        for (int j = 0; j < 4; ++j) s += ia[j] * mi_w1[j * 32 + o];
        m1[o] = geluf(s);
    }
    float l0 = mi_b2[0], l1 = mi_b2[1];
#pragma unroll
    for (int j = 0; j < 32; ++j) {
        l0 += m1[j] * mi_w2[j * 2 + 0];
        l1 += m1[j] * mi_w2[j * 2 + 1];
    }
    float mx = fmaxf(l0, l1);
    float e0 = __expf(l0 - mx), e1 = __expf(l1 - mx);
    float inv = 1.0f / (e0 + e1);
    wImg[r]  = (e0 * inv) * qatt;
    wText[r] = (e1 * inv) * qatt;
}

// ---------------------------------------------------------------------------
// Elementwise modality mixing -> final_img / final_text, stored f32 (needed by
// compensate epilogue) and bf16 (GEMM A operand).
// ---------------------------------------------------------------------------
__global__ void mixk(const float* __restrict__ img, const float* __restrict__ txt,
                     const float* __restrict__ eimg, const float* __restrict__ etxt,
                     const float* __restrict__ quality, const int* __restrict__ missing,
                     float* __restrict__ fImgF, float* __restrict__ fTxtF,
                     unsigned short* __restrict__ fImgH, unsigned short* __restrict__ fTxtH,
                     int B) {
    long long idx = (long long)blockIdx.x * blockDim.x + threadIdx.x;
    if (idx >= (long long)B * HD) return;
    int r = (int)(idx >> 9);
    float ii = quality[(size_t)r * 11 + 6];
    float ti = quality[(size_t)r * 11 + 7];
    float ja = quality[(size_t)r * 11 + 8];   // img_auth
    float ta = quality[(size_t)r * 11 + 9];   // text_auth
    int   m  = missing[r];

    float ratio = ii / (ii + ti + 1e-8f);
    float si0 = ratio > 0.6f ? 1.1f : (ratio < 0.4f ? 0.9f : 1.0f);
    float st0 = ratio > 0.6f ? 0.9f : (ratio < 0.4f ? 1.1f : 1.0f);

    float vi = img[idx], vt = txt[idx];
    float fi, ft;
    if (m == 0) {
        fi = vi * si0;
        ft = vt * st0;
    } else if (m == 1) {
        fi = vi;
        ft = ta * 0.3f * vt + (1.0f - ta) * ti * etxt[idx];
    } else {
        fi = ja * 0.3f * vi + (1.0f - ja) * ii * eimg[idx];
        ft = vt;
    }
    fImgF[idx] = fi;
    fTxtF[idx] = ft;
    fImgH[idx] = f2bf(fi);
    fTxtH[idx] = f2bf(ft);
}

// ---------------------------------------------------------------------------
// Generic 512-K bf16 WMMA GEMM with fused epilogues. gridDim.z = 2 branches.
// WG tile 128x128, 8 waves of wave32, wave tile 64x32 (4x2 WMMA tiles).
// mode 0: D = gelu(acc + bias + difficulty*w1last)           -> bf16
// mode 1: s = sigmoid(acc + bias); D = feat*(1 + diff*s)     -> bf16
// mode 2: D = acc + bias                                     -> bf16
// mode 3: out[row, colOff+col] = (acc + bias) * wRow[row]    -> f32
// ---------------------------------------------------------------------------
__global__ __launch_bounds__(256) void gemm512_stage(
    const unsigned short* __restrict__ A0, const unsigned short* __restrict__ A1,
    const unsigned short* __restrict__ Wp,
    const float* __restrict__ bias,
    const float* __restrict__ w1last,
    const float* __restrict__ quality,
    const float* __restrict__ feat0, const float* __restrict__ feat1,
    const float* __restrict__ wRow0, const float* __restrict__ wRow1,
    unsigned short* __restrict__ Out0, unsigned short* __restrict__ Out1,
    float* __restrict__ OutF,
    int colOff0, int colOff1, int mode, int B) {

    const int z = blockIdx.z;
    const unsigned short* A   = z ? A1 : A0;
    const float*          fea = z ? feat1 : feat0;
    const float*          wRw = z ? wRow1 : wRow0;
    unsigned short*       Out = z ? Out1 : Out0;
    const int          colOff = z ? colOff1 : colOff0;

    const int lane = threadIdx.x & 31;
    const int wid  = threadIdx.x >> 5;
    const int half = lane >> 4;
    const int lm   = lane & 15;

    const int blockRow = blockIdx.x * 128;
    const int blockCol = blockIdx.y * 128;
    const int waveM = (wid >> 2) * 64;      // waves arranged 2 (M) x 4 (N)
    const int waveN = (wid & 3) * 32;

    const v8f zero = {0.f, 0.f, 0.f, 0.f, 0.f, 0.f, 0.f, 0.f};
    v8f acc[4][2];
#pragma unroll
    for (int i = 0; i < 4; ++i)
#pragma unroll
        for (int j = 0; j < 2; ++j) acc[i][j] = zero;

    union AF { v16bf v; v8bf h[2]; };

    for (int kb = 0; kb < HD / 32; ++kb) {
        // A fragments: lanes 0-15 hold K = kb*32 + {0..15},
        //              lanes 16-31 hold K = kb*32 + {16..31}  (matches packed B)
        AF a[4];
#pragma unroll
        for (int mt = 0; mt < 4; ++mt) {
            int row = blockRow + waveM + mt * 16 + lm;
            if (row >= B) row = B - 1;
            const unsigned short* ap = A + (size_t)row * HD + kb * 32 + half * 16;
            a[mt].h[0] = *(const v8bf*)ap;
            a[mt].h[1] = *(const v8bf*)(ap + 8);
        }
        // B fragments: pre-packed, 32 contiguous bytes per lane
        v16bf b[2];
#pragma unroll
        for (int nf = 0; nf < 2; ++nf) {
            int ntG = blockIdx.y * 8 + (wid & 3) * 2 + nf;
            const unsigned short* bp = Wp + (((size_t)kb * (HD / 16) + ntG) * 32 + lane) * 16;
            b[nf] = *(const v16bf*)bp;
        }
#pragma unroll
        for (int mt = 0; mt < 4; ++mt)
#pragma unroll
            for (int nf = 0; nf < 2; ++nf)
                acc[mt][nf] = __builtin_amdgcn_wmma_f32_16x16x32_bf16(
                    false, a[mt].v, false, b[nf], (short)0, acc[mt][nf],
                    false, false);
    }

    // Epilogue. C/D layout: VGPR r -> row = r + 8*half, col = lane%16.
#pragma unroll
    for (int mt = 0; mt < 4; ++mt) {
#pragma unroll
        for (int nf = 0; nf < 2; ++nf) {
#pragma unroll
            for (int r = 0; r < 8; ++r) {
                int row = blockRow + waveM + mt * 16 + half * 8 + r;
                int col = blockCol + waveN + nf * 16 + lm;
                if (row >= B) continue;
                float v = acc[mt][nf][r];
                if (mode == 0) {
                    float d = quality[(size_t)row * 11 + 10];
                    v = geluf(v + bias[col] + d * w1last[col]);
                    Out[(size_t)row * HD + col] = f2bf(v);
                } else if (mode == 1) {
                    float d = quality[(size_t)row * 11 + 10];
                    float s = sigm(v + bias[col]);
                    float f = fea[(size_t)row * HD + col];
                    Out[(size_t)row * HD + col] = f2bf(f * (1.0f + d * s));
                } else if (mode == 2) {
                    Out[(size_t)row * HD + col] = f2bf(v + bias[col]);
                } else {
                    OutF[(size_t)row * 1024 + colOff + col] = (v + bias[col]) * wRw[row];
                }
            }
        }
    }
}

// ---------------------------------------------------------------------------
extern "C" void kernel_launch(void* const* d_in, const int* in_sizes, int n_in,
                              void* d_out, int out_size, void* d_ws, size_t ws_size,
                              hipStream_t stream) {
    const float* image_feat = (const float*)d_in[0];
    const float* text_feat  = (const float*)d_in[1];
    const float* eimg_feat  = (const float*)d_in[2];
    const float* etxt_feat  = (const float*)d_in[3];
    const float* quality    = (const float*)d_in[4];
    const int*   missing    = (const int*)  d_in[5];
    const float* qa_w1 = (const float*)d_in[6],  *qa_b1 = (const float*)d_in[7];
    const float* qa_w2 = (const float*)d_in[8],  *qa_b2 = (const float*)d_in[9];
    const float* qa_w3 = (const float*)d_in[10], *qa_b3 = (const float*)d_in[11];
    const float* mi_w1 = (const float*)d_in[12], *mi_b1 = (const float*)d_in[13];
    const float* mi_w2 = (const float*)d_in[14], *mi_b2 = (const float*)d_in[15];
    const float* dc_w1 = (const float*)d_in[16], *dc_b1 = (const float*)d_in[17];
    const float* dc_w2 = (const float*)d_in[18], *dc_b2 = (const float*)d_in[19];
    const float* wv = (const float*)d_in[20], *bv = (const float*)d_in[21];
    const float* wo = (const float*)d_in[22], *bo = (const float*)d_in[23];

    const int B = in_sizes[0] / HD;

    char* w = (char*)d_ws;
    auto alloc = [&](size_t bytes) -> void* {
        void* p = (void*)w;
        w += (bytes + 255) & ~(size_t)255;
        return p;
    };
    float* finalImgF = (float*)alloc((size_t)B * HD * 4);
    float* finalTxtF = (float*)alloc((size_t)B * HD * 4);
    float* wImg = (float*)alloc((size_t)B * 4);
    float* wTxt = (float*)alloc((size_t)B * 4);
    unsigned short* bufA_img = (unsigned short*)alloc((size_t)B * HD * 2);
    unsigned short* bufA_txt = (unsigned short*)alloc((size_t)B * HD * 2);
    unsigned short* bufB_img = (unsigned short*)alloc((size_t)B * HD * 2);
    unsigned short* bufB_txt = (unsigned short*)alloc((size_t)B * HD * 2);
    unsigned short* pW1 = (unsigned short*)alloc((size_t)HD * HD * 2);
    unsigned short* pW2 = (unsigned short*)alloc((size_t)HD * HD * 2);
    unsigned short* pWv = (unsigned short*)alloc((size_t)HD * HD * 2);
    unsigned short* pWo = (unsigned short*)alloc((size_t)HD * HD * 2);
    const float* w1last = dc_w1 + (size_t)HD * HD;  // 513th row of dc_w1

    // 1) repack weights into WMMA B-fragment layout (first 512 rows of dc_w1)
    int packT = (HD / 32) * (HD / 16) * 32;
    int packG = (packT + 255) / 256;
    pack_weight<<<packG, 256, 0, stream>>>(dc_w1, pW1, HD, HD);
    pack_weight<<<packG, 256, 0, stream>>>(dc_w2, pW2, HD, HD);
    pack_weight<<<packG, 256, 0, stream>>>(wv,    pWv, HD, HD);
    pack_weight<<<packG, 256, 0, stream>>>(wo,    pWo, HD, HD);

    // 2) per-row tiny MLPs
    rowprep<<<(B + 255) / 256, 256, 0, stream>>>(quality, qa_w1, qa_b1, qa_w2, qa_b2,
                                                 qa_w3, qa_b3, mi_w1, mi_b1, mi_w2, mi_b2,
                                                 wImg, wTxt, B);

    // 3) modality mixing
    long long nElem = (long long)B * HD;
    mixk<<<(int)((nElem + 255) / 256), 256, 0, stream>>>(
        image_feat, text_feat, eimg_feat, etxt_feat, quality, missing,
        finalImgF, finalTxtF, bufA_img, bufA_txt, B);

    dim3 gg((B + 127) / 128, HD / 128, 2), bb(256);

    // 4) stage 1: h = gelu(final @ dc_w1[:512] + diff*dc_w1[512] + b1)
    gemm512_stage<<<gg, bb, 0, stream>>>(bufA_img, bufA_txt, pW1, dc_b1, w1last, quality,
                                         nullptr, nullptr, nullptr, nullptr,
                                         bufB_img, bufB_txt, nullptr, 0, 0, 0, B);
    // 5) stage 2: comp = final * (1 + diff * sigmoid(h @ dc_w2 + b2))
    gemm512_stage<<<gg, bb, 0, stream>>>(bufB_img, bufB_txt, pW2, dc_b2, nullptr, quality,
                                         finalImgF, finalTxtF, nullptr, nullptr,
                                         bufA_img, bufA_txt, nullptr, 0, 0, 1, B);
    // 6) stage 3: t = comp @ wv + bv
    gemm512_stage<<<gg, bb, 0, stream>>>(bufA_img, bufA_txt, pWv, bv, nullptr, nullptr,
                                         nullptr, nullptr, nullptr, nullptr,
                                         bufB_img, bufB_txt, nullptr, 0, 0, 2, B);
    // 7) stage 4: attended = t @ wo + bo; scale & scatter into out.
    //    img chain (z=0) -> text_attended -> cols [512,1024) * wText
    //    text chain (z=1) -> img_attended -> cols [0,512)    * wImg
    gemm512_stage<<<gg, bb, 0, stream>>>(bufB_img, bufB_txt, pWo, bo, nullptr, nullptr,
                                         nullptr, nullptr, wTxt, wImg,
                                         nullptr, nullptr, (float*)d_out, 512, 0, 3, B);
}
